// MultiHeadSelfAttention_5549097747091
// MI455X (gfx1250) — matrix operand order repaired and checked by reading, compiled, not run on hardware
//
#include <hip/hip_runtime.h>
#include <hip/hip_bf16.h>
#include <math.h>
#include <stdint.h>

typedef __bf16 bf16;
typedef __attribute__((ext_vector_type(16))) __bf16 v16bf;
typedef __attribute__((ext_vector_type(8)))  float  v8f;
typedef __attribute__((ext_vector_type(4)))  int    v4i;
typedef __attribute__((ext_vector_type(4)))  uint32_t v4u;

#define N_B   8
#define C_DIM 256
#define L_DIM 1024
#define E_DIM 512
#define HC    512
#define HEADS 8
#define DHEAD 64
#define O3    1536

// ---------------------------------------------------------------- CDNA5 async global->LDS staging
#if defined(__has_builtin)
#  if __has_builtin(__builtin_amdgcn_global_load_async_to_lds_b128)
#    define HAS_ASYNC_LDS 1
#  endif
#endif
#ifndef HAS_ASYNC_LDS
#  define HAS_ASYNC_LDS 0
#endif

static __device__ __forceinline__ void stage16(const bf16* g, bf16* l) {
#if HAS_ASYNC_LDS
  __builtin_amdgcn_global_load_async_to_lds_b128(
      (__attribute__((address_space(1))) v4i*)(bf16*)g,
      (__attribute__((address_space(3))) v4i*)l, 0, 0);
#else
  *(v4u*)l = *(const v4u*)g;
#endif
}

static __device__ __forceinline__ void stage_wait() {
#if HAS_ASYNC_LDS
#  if defined(__has_builtin) && __has_builtin(__builtin_amdgcn_s_wait_asynccnt)
  __builtin_amdgcn_s_wait_asynccnt(0);
#  else
  asm volatile("s_wait_asynccnt 0" ::: "memory");
#  endif
#endif
}

static __device__ __forceinline__ v8f wmma_bf16(v16bf a, v16bf b, v8f c) {
  return __builtin_amdgcn_wmma_f32_16x16x32_bf16(false, a, false, b, (short)0, c, false, false);
}

static __device__ __forceinline__ float block_sum_256(float v, float* red) {
  #pragma unroll
  for (int m = 16; m >= 1; m >>= 1) v += __shfl_xor(v, m, 32);
  const int wid = threadIdx.x >> 5;
  __syncthreads();
  if ((threadIdx.x & 31) == 0) red[wid] = v;
  __syncthreads();
  float t = 0.f;
  #pragma unroll
  for (int i = 0; i < 8; ++i) t += red[i];
  return t;
}

// ---------------------------------------------------------------- style = w @ mod_w + 1
__global__ __launch_bounds__(256)
void style_kernel(const float* __restrict__ w, const float* __restrict__ mod_w,
                  float* __restrict__ style) {
  const int n = blockIdx.x, c = threadIdx.x;
  float acc = 1.0f;
  #pragma unroll 8
  for (int e = 0; e < E_DIM; ++e) acc += w[n * E_DIM + e] * mod_w[e * C_DIM + c];
  style[n * C_DIM + c] = acc;
}

// ---------------------------------------------------------------- mp_weight(qkv_w) -> bf16
__global__ __launch_bounds__(256)
void wq_prep_kernel(const float* __restrict__ qkv_w, bf16* __restrict__ Wq) {
  __shared__ float red[8];
  const int row = blockIdx.x, t = threadIdx.x;
  const float v = qkv_w[(size_t)row * C_DIM + t];
  const float mean = block_sum_256(v, red) * (1.0f / 256.0f);
  const float d = v - mean;
  const float n2 = block_sum_256(d * d, red);
  const float scale = 1.0f / ((sqrtf(n2) + 1e-8f) * 16.0f);   // /sqrt(256)
  Wq[(size_t)row * C_DIM + t] = (bf16)(d * scale);
}

// ---------------------------------------------------------------- mp_weight(proj_w)*ResidualGain -> bf16
__global__ __launch_bounds__(256)
void wp_prep_kernel(const float* __restrict__ proj_w, const float* __restrict__ rg,
                    bf16* __restrict__ Wp) {
  __shared__ float red[8];
  const int row = blockIdx.x, t = threadIdx.x;
  const float v0 = proj_w[(size_t)row * HC + t];
  const float v1 = proj_w[(size_t)row * HC + t + 256];
  const float mean = block_sum_256(v0 + v1, red) * (1.0f / 512.0f);
  const float d0 = v0 - mean, d1 = v1 - mean;
  const float n2 = block_sum_256(d0 * d0 + d1 * d1, red);
  const float scale = rg[row] / ((sqrtf(n2) + 1e-8f) * sqrtf(512.0f));
  Wp[(size_t)row * HC + t]       = (bf16)(d0 * scale);
  Wp[(size_t)row * HC + t + 256] = (bf16)(d1 * scale);
}

// ---------------------------------------------------------------- xm = x * style -> bf16
__global__ __launch_bounds__(256)
void xm_kernel(const float* __restrict__ x, const float* __restrict__ style,
               bf16* __restrict__ xm) {
  const size_t i = (size_t)blockIdx.x * 256 + threadIdx.x;   // N*C*L = 2M
  const int c = (int)((i >> 10) & 255);
  const int n = (int)(i >> 18);
  xm[i] = (bf16)(x[i] * style[n * C_DIM + c]);
}

// ---------------------------------------------------------------- y = Wq @ xm  (per n: 1536x256x1024)
// Block tile 128(M) x 128(N), kc=32; 8 waves, one 16x128 strip each (8 accumulators).
__global__ __launch_bounds__(256)
void qkv_gemm_kernel(const bf16* __restrict__ Wq, const bf16* __restrict__ xm,
                     const float* __restrict__ qkv_b, const float* __restrict__ gain,
                     float* __restrict__ y) {
  __shared__ bf16 Ash[128 * 32];      // [row][k]
  __shared__ bf16 Bsh[128 * 32];      // [col][k] (transposed)
  const int tid = threadIdx.x;
  const int w = tid >> 5, lane = tid & 31, lh = lane & 15, half = lane >> 4;
  const int kb = half ? 16 : 0;
  const int n = blockIdx.z, m0 = blockIdx.y * 128, l0 = blockIdx.x * 128;

  v8f c[8];
  #pragma unroll
  for (int ct = 0; ct < 8; ++ct) c[ct] = {};

  for (int k0 = 0; k0 < C_DIM; k0 += 32) {
    __syncthreads();
    #pragma unroll
    for (int p = 0; p < 2; ++p) {            // 128x32 A tile, async 16B chunks
      const int cid = p * 256 + tid;         // 512 chunks
      const int row = cid >> 2, part = cid & 3;
      stage16(Wq + (size_t)(m0 + row) * C_DIM + k0 + part * 8, Ash + cid * 8);
    }
    #pragma unroll
    for (int p = 0; p < 16; ++p) {           // 32x128 B tile, stored transposed [col][k]
      const int idx = p * 256 + tid;
      const int kk = idx >> 7, cc = idx & 127;
      Bsh[cc * 32 + kk] = xm[((size_t)n * C_DIM + k0 + kk) * L_DIM + l0 + cc];
    }
    stage_wait();
    __syncthreads();

    v16bf a;
    const bf16* Ar = &Ash[(w * 16 + lh) * 32];
    #pragma unroll
    for (int i = 0; i < 8; ++i) { a[i] = Ar[kb + i]; a[i + 8] = Ar[kb + 8 + i]; }
    #pragma unroll
    for (int ct = 0; ct < 8; ++ct) {
      const v16bf b = *(const v16bf*)&Bsh[(ct * 16 + lh) * 32 + half * 16];
      c[ct] = wmma_bf16(a, b, c[ct]);
    }
  }
  #pragma unroll
  for (int ct = 0; ct < 8; ++ct)
    #pragma unroll
    for (int r = 0; r < 8; ++r) {
      const int row = m0 + w * 16 + r + half * 8;
      const int col = l0 + ct * 16 + lh;
      y[((size_t)n * O3 + row) * L_DIM + col] = (c[ct][r] + qkv_b[row]) * gain[row];
    }
}

// ---------------------------------------------------------------- RoPE + L2 norm; one wave per 64-wide row
__global__ __launch_bounds__(256)
void rope_norm_kernel(const float* __restrict__ y, bf16* __restrict__ qn,
                      bf16* __restrict__ kn, bf16* __restrict__ vv) {
  const int wid = threadIdx.x >> 5, lane = threadIdx.x & 31;
  const int row = blockIdx.x * 8 + wid;         // n*H*L rows
  const int l = row & 1023, h = (row >> 10) & 7, n = row >> 13;
  const int ih = l >> 5, iw = l & 31;
  const float fr = __powf(10000.0f, -(float)(lane & 15) * (1.0f / 16.0f));
  const float ang = (lane < 16 ? (float)ih : (float)iw) * fr;
  float sn, cs;
  __sincosf(ang, &sn, &cs);

  const size_t obase = ((size_t)n * O3) * L_DIM + l;
  const size_t dst = (((size_t)(n * HEADS + h)) * L_DIM + l) * DHEAD;
  const int j2 = 2 * lane;

  {  // q: o = h*64 + d, A_SCALE folded in
    const size_t b = obase + (size_t)(h * DHEAD + j2) * L_DIM;
    const float x1 = y[b], x2 = y[b + L_DIM];
    const float r1 = x1 * cs - x2 * sn, r2 = x1 * sn + x2 * cs;
    float ss = r1 * r1 + r2 * r2;
    #pragma unroll
    for (int m = 1; m < 32; m <<= 1) ss += __shfl_xor(ss, m, 32);
    const float inv = 8.0f / (sqrtf(ss) + 1e-8f);
    qn[dst + j2] = (bf16)(r1 * inv);
    qn[dst + j2 + 1] = (bf16)(r2 * inv);
  }
  {  // k: o = 512 + h*64 + d
    const size_t b = obase + (size_t)(HC + h * DHEAD + j2) * L_DIM;
    const float x1 = y[b], x2 = y[b + L_DIM];
    const float r1 = x1 * cs - x2 * sn, r2 = x1 * sn + x2 * cs;
    float ss = r1 * r1 + r2 * r2;
    #pragma unroll
    for (int m = 1; m < 32; m <<= 1) ss += __shfl_xor(ss, m, 32);
    const float inv = 1.0f / (sqrtf(ss) + 1e-8f);
    kn[dst + j2] = (bf16)(r1 * inv);
    kn[dst + j2 + 1] = (bf16)(r2 * inv);
  }
  {  // v: o = 1024 + h*64 + d (no rope/norm)
    const size_t b = obase + (size_t)(2 * HC + h * DHEAD + j2) * L_DIM;
    vv[dst + j2] = (bf16)y[b];
    vv[dst + j2 + 1] = (bf16)y[b + L_DIM];
  }
}

// ---------------------------------------------------------------- flash attention, sink-softmax
__global__ __launch_bounds__(256)
void attn_kernel(const bf16* __restrict__ qn, const bf16* __restrict__ kn,
                 const bf16* __restrict__ vv, const float* __restrict__ sinks,
                 bf16* __restrict__ oat) {
  __shared__ bf16 Ksh[64 * 64];       // [key][d]  natural layout
  __shared__ bf16 Vt[64 * 64];        // [d][key]  transposed
  __shared__ bf16 Psh[8 * 16 * 64];   // per-wave P scratch

  const int tid = threadIdx.x;
  const int w = tid >> 5, lane = tid & 31, lh = lane & 15, half = lane >> 4;
  const int kb = half ? 16 : 0;
  const int n = blockIdx.z, h = blockIdx.y, q0 = blockIdx.x * 128;
  const size_t headbase = ((size_t)(n * HEADS + h)) * L_DIM * DHEAD;

  v16bf aq[2];
  {
    const bf16* qr = qn + headbase + (size_t)(q0 + w * 16 + lh) * DHEAD;
    #pragma unroll
    for (int c = 0; c < 2; ++c)
      #pragma unroll
      for (int i = 0; i < 8; ++i) {
        aq[c][i] = qr[32 * c + kb + i];
        aq[c][i + 8] = qr[32 * c + kb + 8 + i];
      }
  }

  float mrow[8], lrow[8];
  v8f o[4];
  #pragma unroll
  for (int r = 0; r < 8; ++r) { mrow[r] = -1e30f; lrow[r] = 0.0f; }
  #pragma unroll
  for (int dt = 0; dt < 4; ++dt) o[dt] = {};

  for (int kt = 0; kt < 16; ++kt) {
    __syncthreads();
    const size_t tb = headbase + (size_t)kt * 64 * DHEAD;
    #pragma unroll
    for (int p = 0; p < 2; ++p) {            // 64x64 K tile, async 16B chunks
      const int cid = p * 256 + tid;         // 512 chunks
      stage16(kn + tb + cid * 8, Ksh + cid * 8);
    }
    const bf16* Vg = vv + tb;
    #pragma unroll
    for (int p = 0; p < 16; ++p) {           // V transposed to [d][key]
      const int idx = p * 256 + tid;
      Vt[(idx & 63) * 64 + (idx >> 6)] = Vg[idx];
    }
    if (kt + 1 < 16) __builtin_prefetch(kn + tb + 64 * DHEAD, 0, 0);
    stage_wait();
    __syncthreads();

    // S = Q K^T  (16 queries x 64 keys)
    v8f s[4];
    #pragma unroll
    for (int ct = 0; ct < 4; ++ct) {
      const bf16* kr = &Ksh[(ct * 16 + lh) * 64];
      const v16bf b0 = *(const v16bf*)(kr + half * 16);
      const v16bf b1 = *(const v16bf*)(kr + 32 + half * 16);
      v8f acc = {};
      acc = wmma_bf16(aq[0], b0, acc);
      acc = wmma_bf16(aq[1], b1, acc);
      s[ct] = acc;
    }

    // online softmax (row = r + 8*half, striped across 16-lane group)
    float scl[8];
    #pragma unroll
    for (int r = 0; r < 8; ++r) {
      float t = fmaxf(fmaxf(s[0][r], s[1][r]), fmaxf(s[2][r], s[3][r]));
      #pragma unroll
      for (int m = 1; m < 16; m <<= 1) t = fmaxf(t, __shfl_xor(t, m, 16));
      const float mn = fmaxf(mrow[r], t);
      scl[r] = __expf(mrow[r] - mn);
      mrow[r] = mn;
      lrow[r] *= scl[r];
    }
    #pragma unroll
    for (int ct = 0; ct < 4; ++ct)
      #pragma unroll
      for (int r = 0; r < 8; ++r) {
        const float p = __expf(s[ct][r] - mrow[r]);
        lrow[r] += p;
        s[ct][r] = p;
      }
    #pragma unroll
    for (int dt = 0; dt < 4; ++dt)
      #pragma unroll
      for (int r = 0; r < 8; ++r) o[dt][r] *= scl[r];

    // P: C-layout -> LDS -> A-layout (per-wave region, wave-order DS)
    bf16* Pw = Psh + w * 16 * 64;
    #pragma unroll
    for (int ct = 0; ct < 4; ++ct)
      #pragma unroll
      for (int r = 0; r < 8; ++r)
        Pw[(r + half * 8) * 64 + ct * 16 + lh] = (bf16)s[ct][r];

    v16bf ap[2];
    const bf16* Pr = Pw + lh * 64;
    #pragma unroll
    for (int c = 0; c < 2; ++c)
      #pragma unroll
      for (int i = 0; i < 8; ++i) {
        ap[c][i] = Pr[32 * c + kb + i];
        ap[c][i + 8] = Pr[32 * c + kb + 8 + i];
      }

    // O += P @ V
    #pragma unroll
    for (int dt = 0; dt < 4; ++dt) {
      const bf16* vr = &Vt[(dt * 16 + lh) * 64];
      const v16bf b0 = *(const v16bf*)(vr + half * 16);
      const v16bf b1 = *(const v16bf*)(vr + 32 + half * 16);
      o[dt] = wmma_bf16(ap[0], b0, o[dt]);
      o[dt] = wmma_bf16(ap[1], b1, o[dt]);
    }
  }

  // fold in the sink logit and normalize
  const float sv = sinks[h];
  float fs[8];
  #pragma unroll
  for (int r = 0; r < 8; ++r) {
    float ls = lrow[r];
    #pragma unroll
    for (int m = 1; m < 16; m <<= 1) ls += __shfl_xor(ls, m, 16);
    const float mf = fmaxf(mrow[r], sv);
    const float sc = __expf(mrow[r] - mf);
    ls = ls * sc + __expf(sv - mf);
    fs[r] = sc / ls;
  }
  #pragma unroll
  for (int dt = 0; dt < 4; ++dt)
    #pragma unroll
    for (int r = 0; r < 8; ++r) {
      const int lq = q0 + w * 16 + r + half * 8;
      const int d = dt * 16 + lh;
      oat[((size_t)(n * HC) + h * DHEAD + d) * L_DIM + lq] = (bf16)(o[dt][r] * fs[r]);
    }
}

// ---------------------------------------------------------------- out = x + Wp @ O  (per n: 256x512x1024)
__global__ __launch_bounds__(256)
void proj_gemm_kernel(const bf16* __restrict__ Wp, const bf16* __restrict__ oat,
                      const float* __restrict__ x, float* __restrict__ out) {
  __shared__ bf16 Ash[128 * 32];
  __shared__ bf16 Bsh[128 * 32];
  const int tid = threadIdx.x;
  const int w = tid >> 5, lane = tid & 31, lh = lane & 15, half = lane >> 4;
  const int kb = half ? 16 : 0;
  const int n = blockIdx.z, m0 = blockIdx.y * 128, l0 = blockIdx.x * 128;

  v8f c[8];
  #pragma unroll
  for (int ct = 0; ct < 8; ++ct) c[ct] = {};

  for (int k0 = 0; k0 < HC; k0 += 32) {
    __syncthreads();
    #pragma unroll
    for (int p = 0; p < 2; ++p) {
      const int cid = p * 256 + tid;
      const int row = cid >> 2, part = cid & 3;
      stage16(Wp + (size_t)(m0 + row) * HC + k0 + part * 8, Ash + cid * 8);
    }
    #pragma unroll
    for (int p = 0; p < 16; ++p) {
      const int idx = p * 256 + tid;
      const int kk = idx >> 7, cc = idx & 127;
      Bsh[cc * 32 + kk] = oat[((size_t)n * HC + k0 + kk) * L_DIM + l0 + cc];
    }
    stage_wait();
    __syncthreads();

    v16bf a;
    const bf16* Ar = &Ash[(w * 16 + lh) * 32];
    #pragma unroll
    for (int i = 0; i < 8; ++i) { a[i] = Ar[kb + i]; a[i + 8] = Ar[kb + 8 + i]; }
    #pragma unroll
    for (int ct = 0; ct < 8; ++ct) {
      const v16bf b = *(const v16bf*)&Bsh[(ct * 16 + lh) * 32 + half * 16];
      c[ct] = wmma_bf16(a, b, c[ct]);
    }
  }
  #pragma unroll
  for (int ct = 0; ct < 8; ++ct)
    #pragma unroll
    for (int r = 0; r < 8; ++r) {
      const int row = m0 + w * 16 + r + half * 8;
      const int col = l0 + ct * 16 + lh;
      const size_t oi = ((size_t)n * C_DIM + row) * L_DIM + col;
      out[oi] = x[oi] + c[ct][r];
    }
}

// ----------------------------------------------------------------
extern "C" void kernel_launch(void* const* d_in, const int* in_sizes, int n_in,
                              void* d_out, int out_size, void* d_ws, size_t ws_size,
                              hipStream_t stream) {
  (void)in_sizes; (void)n_in; (void)out_size; (void)ws_size;

  const float* x     = (const float*)d_in[0];
  const float* w     = (const float*)d_in[1];
  const float* ig    = (const float*)d_in[2];
  const float* rg    = (const float*)d_in[3];
  const float* qkvw  = (const float*)d_in[4];
  const float* qkvb  = (const float*)d_in[5];
  const float* modw  = (const float*)d_in[6];
  const float* projw = (const float*)d_in[7];
  const float* sinks = (const float*)d_in[8];
  float* out = (float*)d_out;

  char* ws = (char*)d_ws;
  size_t off = 0;
  auto alloc = [&](size_t bytes) -> char* {
    char* p = ws + off;
    off = (off + bytes + 255) & ~(size_t)255;
    return p;
  };

  float* style = (float*)alloc((size_t)N_B * C_DIM * 4);
  bf16* Wq = (bf16*)alloc((size_t)O3 * C_DIM * 2);
  bf16* Wp = (bf16*)alloc((size_t)C_DIM * HC * 2);
  bf16* xm = (bf16*)alloc((size_t)N_B * C_DIM * L_DIM * 2);
  float* y = (float*)alloc((size_t)N_B * O3 * L_DIM * 4);
  bf16* qn = (bf16*)alloc((size_t)N_B * HC * L_DIM * 2);
  bf16* kn = (bf16*)alloc((size_t)N_B * HC * L_DIM * 2);
  bf16* vv = (bf16*)alloc((size_t)N_B * HC * L_DIM * 2);
  bf16* oat = (bf16*)alloc((size_t)N_B * HC * L_DIM * 2);

  style_kernel<<<N_B, 256, 0, stream>>>(w, modw, style);
  wq_prep_kernel<<<O3, 256, 0, stream>>>(qkvw, Wq);
  wp_prep_kernel<<<C_DIM, 256, 0, stream>>>(projw, rg, Wp);
  xm_kernel<<<(N_B * C_DIM * L_DIM) / 256, 256, 0, stream>>>(x, style, xm);
  qkv_gemm_kernel<<<dim3(L_DIM / 128, O3 / 128, N_B), 256, 0, stream>>>(Wq, xm, qkvb, ig, y);
  rope_norm_kernel<<<(N_B * HEADS * L_DIM) / 8, 256, 0, stream>>>(y, qn, kn, vv);
  attn_kernel<<<dim3(L_DIM / 128, HEADS, N_B), 256, 0, stream>>>(qn, kn, vv, sinks, oat);
  proj_gemm_kernel<<<dim3(L_DIM / 128, C_DIM / 128, N_B), 256, 0, stream>>>(Wp, oat, x, out);
}
